// CausalSelfAttention_87926570483981
// MI455X (gfx1250) — matrix operand order repaired and checked by reading, not compiled, not run on hardware
//
#include <hip/hip_runtime.h>
#include <stdint.h>

// CDNA5 / gfx1250: wave32, WMMA 16x16x32 bf16 -> f32 accum, TDM staging,
// non-temporal hints on the streaming 512MB attn tensor.
typedef __attribute__((ext_vector_type(16))) __bf16       v16bf;
typedef __attribute__((ext_vector_type(8)))  float        v8f;
typedef __attribute__((ext_vector_type(4)))  float        v4f;   // clang vector (NT-builtin friendly)
typedef __attribute__((ext_vector_type(4)))  unsigned int v4u;
typedef __attribute__((ext_vector_type(8)))  int          v8i;
typedef __attribute__((ext_vector_type(4)))  int          v4i;

#define BB 2
#define NN 2048
#define DD 1024
#define HH 16
#define DH 64
#define E3 192  // 3*DH

// Native f32->bf16 conversion (RNE): let the backend pick v_cvt_pk_bf16_f32
// instead of the 3-VALU manual rounding sequence.
static __device__ __forceinline__ __bf16 f2bf(float f) { return (__bf16)f; }

static __device__ __forceinline__ v8f wmma_bf16(v16bf a, v16bf b, v8f c) {
  // (neg_a, A, neg_b, B, c_mod, C, reuse_a, reuse_b)
  return __builtin_amdgcn_wmma_f32_16x16x32_bf16(false, a, false, b, (short)0, c, false, false);
}

template <int O>
static __device__ __forceinline__ void cvt8(v16bf& d, v4f a, v4f b) {
  d[O + 0] = f2bf(a.x); d[O + 1] = f2bf(a.y); d[O + 2] = f2bf(a.z); d[O + 3] = f2bf(a.w);
  d[O + 4] = f2bf(b.x); d[O + 5] = f2bf(b.y); d[O + 6] = f2bf(b.z); d[O + 7] = f2bf(b.w);
}

// A-fragment 16x32 bf16: per lane, K = {half*16 .. half*16+15}
// -> one contiguous 16-float run => 4x b128 loads.
static __device__ __forceinline__ v16bf load_a_bf(const float* row, int half) {
  const v4f* p0 = (const v4f*)(row + half * 16);
  const v4f* p1 = (const v4f*)(row + half * 16 + 8);
  v16bf d;
  cvt8<0>(d, p0[0], p0[1]);
  cvt8<8>(d, p1[0], p1[1]);
  return d;
}

// B-fragment when source K-dim is contiguous: per lane, 16 contiguous floats.
static __device__ __forceinline__ v16bf load_b_bf(const float* p) {
  const v4f* q = (const v4f*)p;
  v16bf d;
  cvt8<0>(d, q[0], q[1]);
  cvt8<8>(d, q[2], q[3]);
  return d;
}

static __device__ __forceinline__ v4f norm4(v4f s, float m, float il) {
  const float NEG = -__builtin_inff();
  v4f r;
  r.x = (s.x > NEG) ? __expf(s.x - m) * il : 0.0f;
  r.y = (s.y > NEG) ? __expf(s.y - m) * il : 0.0f;
  r.z = (s.z > NEG) ? __expf(s.z - m) * il : 0.0f;
  r.w = (s.w > NEG) ? __expf(s.w - m) * il : 0.0f;
  return r;
}

// ---- TDM: 1D contiguous tile (nelem f32) global -> LDS (CDNA5 ISA, D#) ---
static __device__ __forceinline__ void tdm_load_1d(uint32_t lds_byte,
                                                   const float* gptr, int nelem) {
  v4u g0 = {};
  g0[0] = 1u;                                    // count=1, user descriptor
  g0[1] = lds_byte;                              // lds_addr (bytes)
  uint64_t ga = (uint64_t)(size_t)gptr;          // global tile start
  g0[2] = (uint32_t)ga;
  g0[3] = (uint32_t)((ga >> 32) & 0x01FFFFFFu) | 0x80000000u;  // type=2 "image"
  v8i g1 = {};
  uint32_t td0 = (uint32_t)nelem;
  g1[0] = (int)(2u << 16);                       // data_size = 2 (4 bytes)
  g1[1] = (int)((td0 & 0xFFFFu) << 16);          // tensor_dim0[15:0]
  g1[2] = (int)(((td0 >> 16) & 0xFFFFu) | (1u << 16));  // tensor_dim0 hi, dim1=1
  g1[3] = (int)(td0 << 16);                      // tile_dim0 = nelem (1D tile)
  g1[5] = (int)td0;                              // tensor_dim0_stride lo32
  v4i z4 = {};
#if defined(__clang_major__) && (__clang_major__ >= 23)
  v8i z8 = {};
  __builtin_amdgcn_tensor_load_to_lds(g0, g1, z4, z4, z8, 0);
#else
  __builtin_amdgcn_tensor_load_to_lds(g0, g1, z4, z4, 0);
#endif
}

// ---------------- K1: per-head QKV projection -------------------------------
__global__ void qkv_proj_kernel(const float* __restrict__ x,
                                const float* __restrict__ Wqkv,
                                const float* __restrict__ bqkv,
                                float* __restrict__ Kbuf,
                                float* __restrict__ Qbuf,
                                float* __restrict__ Vbuf) {
  const int lane = threadIdx.x & 31;
  const int half = lane >> 4, l16 = lane & 15;
  const int row0 = blockIdx.x * 16;   // over B*N = 4096
  const int col0 = blockIdx.y * 16;   // over 192
  const int h = blockIdx.z;
  const float* W = Wqkv + (size_t)h * DD * E3;
  v8f c = {};
  for (int k0 = 0; k0 < DD; k0 += 32) {
    v16bf a = load_a_bf(x + (size_t)(row0 + l16) * DD + k0, half);
    v16bf b;
#pragma unroll
    for (int e = 0; e < 16; ++e)  // stride-E3 walk; coalesced across lanes
      b[e] = f2bf(W[(size_t)(k0 + half * 16 + e) * E3 + col0 + l16]);
    c = wmma_bf16(a, b, c);
  }
  const int eg = col0 + l16;
  const int chunk = col0 / DH;              // 0=k,1=q,2=v (tile never crosses)
  const int dh = eg - chunk * DH;
  const float bias = bqkv[h * E3 + eg];
  float* dst = (chunk == 0) ? Kbuf : (chunk == 1) ? Qbuf : Vbuf;
#pragma unroll
  for (int r = 0; r < 8; ++r) {
    int gr = row0 + half * 8 + r;
    int bb = gr / NN, n = gr % NN;
    dst[((size_t)(bb * HH + h) * NN + n) * DH + dh] = c[r] + bias;
  }
}

// ---------------- K2: S = scale * Q K^T with causal mask (raw scores) -------
__global__ void scores_kernel(const float* __restrict__ Qbuf,
                              const float* __restrict__ Kbuf,
                              float* __restrict__ attn) {
  const int lane = threadIdx.x & 31;
  const int half = lane >> 4, l16 = lane & 15;
  const int row0 = blockIdx.x * 16;   // q tile
  const int col0 = blockIdx.y * 16;   // k tile
  const int bh = blockIdx.z;          // b*H + h
  float* S = attn + (size_t)bh * NN * NN;
  const float NEG = -__builtin_inff();
  if (col0 > row0 + 15) {             // fully masked tile: no compute
#pragma unroll
    for (int r = 0; r < 8; ++r) {
      int q = row0 + half * 8 + r;
      S[(size_t)q * NN + col0 + l16] = NEG;
    }
    return;
  }
  const size_t base = (size_t)bh * NN * DH;
  v8f c = {};
#pragma unroll
  for (int d0 = 0; d0 < DH; d0 += 32) {
    v16bf a = load_a_bf(Qbuf + base + (size_t)(row0 + l16) * DH + d0, half);
    v16bf b = load_b_bf(Kbuf + base + (size_t)(col0 + l16) * DH + d0 + half * 16);
    c = wmma_bf16(a, b, c);
  }
  const float scale = 0.125f;  // 1/sqrt(64)
#pragma unroll
  for (int r = 0; r < 8; ++r) {
    int q = row0 + half * 8 + r;
    int kc = col0 + l16;
    float s = c[r] * scale;
    if (kc > q) s = NEG;
    S[(size_t)q * NN + kc] = s;
  }
}

// ---------------- K3: per-row online softmax stats (m, l) -------------------
__global__ void softmax_stats_kernel(const float* __restrict__ attn,
                                     float* __restrict__ mrow,
                                     float* __restrict__ lrow) {
  const int lane = threadIdx.x & 31;
  const int q = blockIdx.x;
  const int bh = blockIdx.y;
  const float* Srow = attn + (size_t)bh * NN * NN + (size_t)q * NN;
  const float NEG = -__builtin_inff();
  float m = NEG, l = 0.f;
  for (int k0 = lane * 4; k0 < NN; k0 += 128) {
    v4f v = __builtin_nontemporal_load((const v4f*)(Srow + k0));  // read-once stream
    float vv[4] = {v.x, v.y, v.z, v.w};
#pragma unroll
    for (int j = 0; j < 4; ++j) {
      float vj = vv[j];
      if (!(vj > NEG)) continue;        // masked
      float nm = fmaxf(m, vj);
      l = l * __expf(m - nm) + __expf(vj - nm);
      m = nm;
    }
  }
#pragma unroll
  for (int off = 16; off > 0; off >>= 1) {
    float m2 = __shfl_xor(m, off, 32);
    float l2 = __shfl_xor(l, off, 32);
    float nm = fmaxf(m, m2);
    float a1 = (m > NEG) ? l * __expf(m - nm) : 0.f;
    float a2 = (m2 > NEG) ? l2 * __expf(m2 - nm) : 0.f;
    m = nm; l = a1 + a2;
  }
  if (lane == 0) {
    mrow[(size_t)bh * NN + q] = m;
    lrow[(size_t)bh * NN + q] = l;
  }
}

// -------- K4: fused normalize-write attn + P@V, V staged in LDS via TDM -----
// 8 waves/block, each wave owns one 16-row q-tile; wave 0 DMAs the shared
// 32x64 V chunk into LDS with tensor_load_to_lds, synced by TENSORcnt+barrier.
// Final attn values are written non-temporally (never re-read; 512MB stream).
__global__ __launch_bounds__(256) void attn_pv_kernel(
    float* __restrict__ attn, const float* __restrict__ Vbuf,
    const float* __restrict__ mrow, const float* __restrict__ lrow,
    float* __restrict__ sa) {
  const int tid = threadIdx.x;
  const int lane = tid & 31, wave = tid >> 5;
  const int half = lane >> 4, l16 = lane & 15;
  const int row0 = blockIdx.x * 128 + wave * 16;
  const int bh = blockIdx.y;
  __shared__ float Vlds[32 * DH];   // 8 KB chunk of V
  float* S = attn + (size_t)bh * NN * NN;
  const float* V = Vbuf + (size_t)bh * NN * DH;
  const float m_r = mrow[(size_t)bh * NN + row0 + l16];
  const float inv_l = 1.0f / lrow[(size_t)bh * NN + row0 + l16];
  float* Srow = S + (size_t)(row0 + l16) * NN;
  const uint32_t lds_byte = (uint32_t)(size_t)Vlds;  // low 32b of flat = LDS off
  v8f acc[4] = {};
  for (int ct = 0; ct < NN / 32; ++ct) {
    const int kbase = ct * 32;
    if (wave == 0) {
      tdm_load_1d(lds_byte, V + (size_t)kbase * DH, 32 * DH);
      __builtin_amdgcn_s_wait_tensorcnt(0);
    }
    __syncthreads();
    if (kbase > row0 + 15) {           // fully above diagonal: attn = 0
      v4f z = 0.0f;
      __builtin_nontemporal_store(z, (v4f*)(Srow + kbase + half * 8));
      __builtin_nontemporal_store(z, (v4f*)(Srow + kbase + half * 8 + 4));
      __builtin_nontemporal_store(z, (v4f*)(Srow + kbase + 16 + half * 8));
      __builtin_nontemporal_store(z, (v4f*)(Srow + kbase + 16 + half * 8 + 4));
    } else {
      v4f s0 = *(const v4f*)(Srow + kbase + half * 8);
      v4f s1 = *(const v4f*)(Srow + kbase + half * 8 + 4);
      v4f s2 = *(const v4f*)(Srow + kbase + 16 + half * 8);
      v4f s3 = *(const v4f*)(Srow + kbase + 16 + half * 8 + 4);
      s0 = norm4(s0, m_r, inv_l);
      s1 = norm4(s1, m_r, inv_l);
      s2 = norm4(s2, m_r, inv_l);
      s3 = norm4(s3, m_r, inv_l);
      __builtin_nontemporal_store(s0, (v4f*)(Srow + kbase + half * 8));
      __builtin_nontemporal_store(s1, (v4f*)(Srow + kbase + half * 8 + 4));
      __builtin_nontemporal_store(s2, (v4f*)(Srow + kbase + 16 + half * 8));
      __builtin_nontemporal_store(s3, (v4f*)(Srow + kbase + 16 + half * 8 + 4));
      v16bf a;                                        // P tile as A-fragment
      cvt8<0>(a, s0, s1);
      cvt8<8>(a, s2, s3);
#pragma unroll
      for (int t = 0; t < 4; ++t) {                   // dh tiles of 16
        v16bf b;
#pragma unroll
        for (int e = 0; e < 16; ++e)
          b[e] = f2bf(Vlds[((e < 8 ? half * 8 : 8 + half * 8) + e) * DH + t * 16 + l16]);
        acc[t] = wmma_bf16(a, b, acc[t]);
      }
    }
    __syncthreads();  // protect Vlds before next stage
  }
  const int bb = bh / HH, h = bh % HH;
#pragma unroll
  for (int t = 0; t < 4; ++t)
#pragma unroll
    for (int r = 0; r < 8; ++r) {
      int q = row0 + half * 8 + r;
      sa[((size_t)(bb * NN + q)) * DD + h * DH + t * 16 + l16] = acc[t][r];
    }
}

// ---------------- K5: out = sa @ Wproj^T + bproj ---------------------------
__global__ void out_proj_kernel(const float* __restrict__ sa,
                                const float* __restrict__ Wproj,
                                const float* __restrict__ bproj,
                                float* __restrict__ out) {
  const int lane = threadIdx.x & 31;
  const int half = lane >> 4, l16 = lane & 15;
  const int row0 = blockIdx.x * 16;   // over B*N
  const int col0 = blockIdx.y * 16;   // over D
  v8f c = {};
  for (int k0 = 0; k0 < DD; k0 += 32) {
    v16bf a = load_a_bf(sa + (size_t)(row0 + l16) * DD + k0, half);
    v16bf b = load_b_bf(Wproj + (size_t)(col0 + l16) * DD + k0 + half * 16);
    c = wmma_bf16(a, b, c);
  }
  const float bias = bproj[col0 + l16];
#pragma unroll
  for (int r = 0; r < 8; ++r) {
    int gr = row0 + half * 8 + r;
    out[(size_t)gr * DD + col0 + l16] = c[r] + bias;
  }
}

extern "C" void kernel_launch(void* const* d_in, const int* in_sizes, int n_in,
                              void* d_out, int out_size, void* d_ws, size_t ws_size,
                              hipStream_t stream) {
  const float* x     = (const float*)d_in[0];
  const float* Wqkv  = (const float*)d_in[1];
  const float* bqkv  = (const float*)d_in[2];
  const float* Wproj = (const float*)d_in[3];
  const float* bproj = (const float*)d_in[4];

  float* out  = (float*)d_out;                     // [B,N,D]
  float* attn = out + (size_t)BB * NN * DD;        // [B,H,N,N]

  float* ws = (float*)d_ws;
  const size_t qkvsz = (size_t)BB * HH * NN * DH;  // 4,194,304 each
  float* Kbuf = ws;
  float* Qbuf = Kbuf + qkvsz;
  float* Vbuf = Qbuf + qkvsz;
  float* mrow = Vbuf + qkvsz;
  float* lrow = mrow + (size_t)BB * HH * NN;
  float* sa   = lrow + (size_t)BB * HH * NN;       // total ~67.6 MB

  dim3 w32(32, 1, 1);  // one wave32 per block (one 16x16 WMMA tile)
  qkv_proj_kernel<<<dim3((BB * NN) / 16, E3 / 16, HH), w32, 0, stream>>>(
      x, Wqkv, bqkv, Kbuf, Qbuf, Vbuf);
  scores_kernel<<<dim3(NN / 16, NN / 16, BB * HH), w32, 0, stream>>>(
      Qbuf, Kbuf, attn);
  softmax_stats_kernel<<<dim3(NN, BB * HH, 1), w32, 0, stream>>>(
      attn, mrow, lrow);
  attn_pv_kernel<<<dim3(NN / 128, BB * HH, 1), dim3(256, 1, 1), 0, stream>>>(
      attn, Vbuf, mrow, lrow, sa);
  out_proj_kernel<<<dim3((BB * NN) / 16, DD / 16, 1), w32, 0, stream>>>(
      sa, Wproj, bproj, out);
}
